// Net_75230647157021
// MI455X (gfx1250) — matrix operand order repaired and checked by reading, not compiled, not run on hardware
//
#include <hip/hip_runtime.h>
#include <hip/hip_bf16.h>

typedef __attribute__((ext_vector_type(16))) _Float16 v16h;
typedef __attribute__((ext_vector_type(8)))  _Float16 v8h;
typedef __attribute__((ext_vector_type(8)))  float    v8f;

#define WAVES   2
#define SH16    136   // f16 LDS row stride (68 words -> no 64-bank alias)
#define SF32    132   // f32 LDS row stride
#define SMOE    100
#define SSH     72
#define LN_EPS  1e-5f
#define NLAYERS 38

struct KP {
    const float* x;
    float* out;
    const _Float16* w[NLAYERS];
    const float* b[NLAYERS];
    const float* g_stem1; const float* bt_stem1;
    const float* g_stem2; const float* bt_stem2;
    const float* g_b1l1;  const float* bt_b1l1;
    const float* g_b1l2;  const float* bt_b1l2;
    const float* g_b2l1;  const float* bt_b2l1;
    const float* g_b2l2;  const float* bt_b2l2;
    const float* e_lng;   const float* e_lnb;
    const float* g_post;  const float* bt_post;
};

__device__ __forceinline__ float siluf(float v) { return v / (1.0f + __expf(-v)); }

// Per-wave ordering point. All staging buffers are private to one wave and the
// ISA keeps same-wave LDS ops in order, so no s_barrier is needed -- only a
// compiler scheduling/memory fence so ds_store -> ds_load (cross-lane) is not
// reordered or forwarded.
__device__ __forceinline__ void wsync() {
    __builtin_amdgcn_fence(__ATOMIC_RELEASE, "workgroup");
    __builtin_amdgcn_wave_barrier();
    __builtin_amdgcn_fence(__ATOMIC_ACQUIRE, "workgroup");
}

// A fragment (16x32 f16), per CDNA5 ISA 7.12.2:
// lanes 0-15 row M=lane,     K = k0+{0..7} and k0+{16..23}
// lanes 16-31 row M=lane-16, K = k0+{8..15} and k0+{24..31}
__device__ __forceinline__ v16h load_a(const _Float16* s, int lane, int k0, int stride) {
    int row = lane & 15;
    int kb  = k0 + ((lane & 16) ? 8 : 0);
    const _Float16* p = s + row * stride + kb;
    v8h lo = *(const v8h*)(p);
    v8h hi = *(const v8h*)(p + 16);
    return __builtin_shufflevector(lo, hi, 0,1,2,3,4,5,6,7,8,9,10,11,12,13,14,15);
}

// B fragment (32x16 f16): lanes 0-15 col=n0+lane K=k0+{0..7} and k0+{16..23};
// lanes 16-31 col=n0+lane-16 K=k0+{8..15} and k0+{24..31}.  Weights pre-packed
// column-major [Npad][Kpad] so each lane reads two b128 loads.
__device__ __forceinline__ v16h load_b(const _Float16* w, int lane, int k0, int n0, int Kp) {
    int col = n0 + (lane & 15);
    int kb  = k0 + ((lane & 16) ? 8 : 0);
    const _Float16* p = w + col * Kp + kb;
    v8h lo = *(const v8h*)(p);
    v8h hi = *(const v8h*)(p + 16);
    return __builtin_shufflevector(lo, hi, 0,1,2,3,4,5,6,7,8,9,10,11,12,13,14,15);
}

// 16-row tile GEMM: sOut[16][Np] = f16(sIn[16][Kp]) * W + bias  (per-wave).
// A fragments hoisted to registers (<=4 x v16h); bias folded into the WMMA C
// operand (each lane owns one column across its 8 C VGPRs).
__device__ __forceinline__ void wave_gemm(const _Float16* __restrict__ W,
                                          const float* __restrict__ bias,
                                          int Kp, int Np,
                                          const _Float16* sIn, int inStride,
                                          float* sOut, int lane) {
    const int nk = Kp >> 5;
    v16h a[4];
#pragma unroll
    for (int k = 0; k < 4; ++k)
        if (k < nk) a[k] = load_a(sIn, lane, k << 5, inStride);

    const int colo = lane & 15;
    const int rb   = (lane & 16) ? 8 : 0;
    for (int n0 = 0; n0 < Np; n0 += 16) {
        float bv = bias[n0 + colo];
        v8f c = {bv, bv, bv, bv, bv, bv, bv, bv};
#pragma unroll
        for (int k = 0; k < 4; ++k) {
            if (k < nk) {
                v16h b = load_b(W, lane, k << 5, n0, Kp);
                c = __builtin_amdgcn_wmma_f32_16x16x32_f16(false, a[k], false, b,
                                                           (short)0, c, false, false);
            }
        }
#pragma unroll
        for (int i = 0; i < 8; ++i) sOut[(rb + i) * SF32 + (n0 + colo)] = c[i];
    }
}

// LayerNorm over N real columns (half-row per lane + shfl_xor(16) combine),
// optional SiLU, writes f32 and/or f16 outputs.
__device__ __forceinline__ void ew_ln(const float* sIn, int N,
                                      const float* g, const float* bb, int lane,
                                      float* outF32, _Float16* outF16, bool do_silu) {
    int row = lane & 15;
    int c0  = (lane >> 4) * (N >> 1), c1 = c0 + (N >> 1);
    const float* rp = sIn + row * SF32;
    float s = 0.f, s2 = 0.f;
    for (int c = c0; c < c1; ++c) { float v = rp[c]; s += v; s2 += v * v; }
    s  += __shfl_xor(s, 16);
    s2 += __shfl_xor(s2, 16);
    float mean = s / (float)N;
    float var  = s2 / (float)N - mean * mean;
    float inv  = rsqrtf(var + LN_EPS);
    for (int c = c0; c < c1; ++c) {
        float v = (rp[c] - mean) * inv * g[c] + bb[c];
        if (do_silu) v = siluf(v);
        if (outF32) outF32[row * SF32 + c] = v;
        if (outF16) outF16[row * SH16 + c] = (_Float16)v;
    }
}

__device__ __forceinline__ void ew_silu_f16(const float* sIn, int N, int lane,
                                            _Float16* out, int outStride) {
    int row = lane & 15;
    int c0  = (lane >> 4) * (N >> 1), c1 = c0 + (N >> 1);
    for (int c = c0; c < c1; ++c)
        out[row * outStride + c] = (_Float16)siluf(sIn[row * SF32 + c]);
}

// Prefetch a packed f16 weight matrix (bytes) into cache ahead of its GEMM.
__device__ __forceinline__ void prefetch_w(const _Float16* w, int bytes, int lane) {
    const char* p = (const char*)w;
    int lines = bytes >> 8;              // 256B granules
    for (int off = lane; off < lines; off += 32)
        __builtin_prefetch(p + (off << 8), 0, 1);
}

__global__ void conv_w(const float* __restrict__ src, _Float16* __restrict__ dst,
                       int K, int N, int Kp, int Np) {
    int idx = blockIdx.x * blockDim.x + threadIdx.x;
    if (idx >= Kp * Np) return;
    int n = idx / Kp, k = idx % Kp;
    float v = (k < K && n < N) ? src[k * N + n] : 0.f;  // src is [K][N] row-major
    dst[n * Kp + k] = (_Float16)v;                       // dst col-major for B-frag
}

__global__ void conv_b(const float* __restrict__ src, float* __restrict__ dst, int N, int Np) {
    int i = blockIdx.x * blockDim.x + threadIdx.x;
    if (i < Np) dst[i] = (i < N) ? src[i] : 0.f;
}

__global__ __launch_bounds__(32 * WAVES) void fused_net(KP p) {
    __shared__ alignas(16) _Float16 sX_[WAVES][16 * SH16];
    __shared__ alignas(16) _Float16 sH_[WAVES][16 * SH16];
    __shared__ alignas(16) _Float16 sS_[WAVES][16 * SSH];
    __shared__ alignas(16) float    sF_[WAVES][16 * SF32];
    __shared__ float sM_[WAVES][16 * SMOE];
    __shared__ float sG_[WAVES][16 * 8];
    __shared__ float sR_[WAVES][16 * 4];

    const int lane = threadIdx.x & 31;
    const int w    = threadIdx.x >> 5;
    _Float16* X = sX_[w];
    _Float16* H = sH_[w];
    _Float16* S = sS_[w];
    float*    F = sF_[w];
    float*    M = sM_[w];
    float*    G = sG_[w];
    float*    R = sR_[w];

    const long row0 = ((long)blockIdx.x * WAVES + w) * 16;
    const int  row  = lane & 15;
    const int  half = lane >> 4;

    // ---- load & quantize input tile [16][64] (float4 vectorized) ----
    for (int i = lane; i < 16 * 16; i += 32) {
        int r = i >> 4, c4 = (i & 15) << 2;
        float4 v = *(const float4*)(p.x + (row0 + r) * 64 + c4);
        X[r * SH16 + c4 + 0] = (_Float16)v.x;
        X[r * SH16 + c4 + 1] = (_Float16)v.y;
        X[r * SH16 + c4 + 2] = (_Float16)v.z;
        X[r * SH16 + c4 + 3] = (_Float16)v.w;
    }
    wsync();

    // ---- stem ----
    wave_gemm(p.w[0], p.b[0], 64, 128, X, SH16, F, lane); wsync();
    ew_ln(F, 128, p.g_stem1, p.bt_stem1, lane, nullptr, X, true); wsync();
    wave_gemm(p.w[1], p.b[1], 128, 96, X, SH16, F, lane); wsync();
    ew_ln(F, 96, p.g_stem2, p.bt_stem2, lane, nullptr, H, true); wsync();

    // ---- residual blocks ----
    for (int blk = 0; blk < 2; ++blk) {
        const int lw = 2 + blk * 2;
        const float* g1 = blk ? p.g_b2l1 : p.g_b1l1;
        const float* b1 = blk ? p.bt_b2l1 : p.bt_b1l1;
        const float* g2 = blk ? p.g_b2l2 : p.g_b1l2;
        const float* b2 = blk ? p.bt_b2l2 : p.bt_b1l2;
        wave_gemm(p.w[lw], p.b[lw], 96, 128, H, SH16, F, lane); wsync();
        ew_ln(F, 128, g1, b1, lane, nullptr, X, true); wsync();
        wave_gemm(p.w[lw + 1], p.b[lw + 1], 128, 96, X, SH16, F, lane); wsync();
        ew_ln(F, 96, g2, b2, lane, F, nullptr, false); wsync();
        for (int c = half * 48; c < half * 48 + 48; ++c) {
            float v = F[row * SF32 + c] + (float)H[row * SH16 + c];
            H[row * SH16 + c] = (_Float16)siluf(v);
        }
        wsync();
    }

    // ---- gate ----
    wave_gemm(p.w[6], p.b[6], 96, 64, H, SH16, F, lane); wsync();
    ew_silu_f16(F, 64, lane, X, SH16); wsync();
    wave_gemm(p.w[7], p.b[7], 64, 32, X, SH16, F, lane); wsync();
    ew_silu_f16(F, 32, lane, X, SH16); wsync();
    wave_gemm(p.w[8], p.b[8], 32, 16, X, SH16, F, lane); wsync();
    if (lane < 16) {
        float lg[6], mx = -1e30f;
        for (int e = 0; e < 6; ++e) { lg[e] = F[lane * SF32 + e]; mx = fmaxf(mx, lg[e]); }
        float sum = 0.f;
        for (int e = 0; e < 6; ++e) { lg[e] = __expf(lg[e] - mx); sum += lg[e]; }
        float inv = 1.f / sum;
        for (int e = 0; e < 6; ++e) G[lane * 8 + e] = lg[e] * inv;
    }
    wsync();

    // ---- experts (dense, gated accumulation) ----
    for (int c = half * 48; c < half * 48 + 48; ++c) M[row * SMOE + c] = 0.f;
    prefetch_w(p.w[9], 96 * 96 * 2, lane);
    wsync();
    for (int e = 0; e < 6; ++e) {
        wave_gemm(p.w[9 + e], p.b[9 + e], 96, 96, H, SH16, F, lane); wsync();
        prefetch_w(p.w[15 + e], 96 * 96 * 2, lane);
        ew_ln(F, 96, p.e_lng + e * 96, p.e_lnb + e * 96, lane, nullptr, X, true); wsync();
        wave_gemm(p.w[15 + e], p.b[15 + e], 96, 96, X, SH16, F, lane); wsync();
        prefetch_w(p.w[21 + e], 96 * 96 * 2, lane);
        ew_silu_f16(F, 96, lane, X, SH16); wsync();
        wave_gemm(p.w[21 + e], p.b[21 + e], 96, 96, X, SH16, F, lane); wsync();
        if (e < 5) prefetch_w(p.w[10 + e], 96 * 96 * 2, lane);
        {
            float gw = G[row * 8 + e];
            for (int c = half * 48; c < half * 48 + 48; ++c)
                M[row * SMOE + c] += gw * siluf(F[row * SF32 + c]);
        }
        wsync();
    }

    // ---- post + shared ----
    for (int c = half * 48; c < half * 48 + 48; ++c)
        X[row * SH16 + c] = (_Float16)(M[row * SMOE + c] + (float)H[row * SH16 + c]);
    wsync();
    wave_gemm(p.w[27], p.b[27], 96, 96, X, SH16, F, lane); wsync();
    ew_ln(F, 96, p.g_post, p.bt_post, lane, nullptr, X, true); wsync();
    wave_gemm(p.w[28], p.b[28], 96, 64, X, SH16, F, lane); wsync();
    ew_silu_f16(F, 64, lane, S, SSH); wsync();

    // ---- heads: strain, tensile, yield_gap ----
    for (int h = 0; h < 3; ++h) {
        const int lw = 29 + h * 3;
        wave_gemm(p.w[lw], p.b[lw], 64, 64, S, SSH, F, lane); wsync();
        ew_silu_f16(F, 64, lane, X, SH16); wsync();
        wave_gemm(p.w[lw + 1], p.b[lw + 1], 64, 32, X, SH16, F, lane); wsync();
        ew_silu_f16(F, 32, lane, X, SH16); wsync();
        wave_gemm(p.w[lw + 2], p.b[lw + 2], 32, 16, X, SH16, F, lane); wsync();
        if (lane < 16) R[lane * 4 + h] = F[lane * SF32 + 0];
        wsync();
    }

    if (lane < 16) {
        float st = R[lane * 4 + 0];
        float te = R[lane * 4 + 1];
        float gp = R[lane * 4 + 2];
        float sp = (gp > 20.f) ? gp : log1pf(__expf(gp));
        long gr = row0 + lane;
        p.out[gr * 3 + 0] = st;
        p.out[gr * 3 + 1] = te;
        p.out[gr * 3 + 2] = te - sp;
    }
}

extern "C" void kernel_launch(void* const* d_in, const int* in_sizes, int n_in,
                              void* d_out, int out_size, void* d_ws, size_t ws_size,
                              hipStream_t stream) {
    struct WDef { int wi, wo, K, N, Kp, Np, bi, bo, bN; };
    WDef L[NLAYERS];
    int li = 0;
    auto add = [&](int wi, int wo, int K, int N, int Kp, int Np, int bi, int bo, int bN) {
        L[li++] = {wi, wo, K, N, Kp, Np, bi, bo, bN};
    };
    // pytree (sorted-dict) leaf order: idx0 = x, then params leaves
    add(56, 0,  64, 128,  64, 128, 55, 0, 128);  // 0 stem_l1
    add(58, 0, 128,  96, 128,  96, 57, 0,  96);  // 1 stem_l2
    add( 2, 0,  96, 128,  96, 128,  1, 0, 128);  // 2 b1.fc1
    add( 4, 0, 128,  96, 128,  96,  3, 0,  96);  // 3 b1.fc2
    add(10, 0,  96, 128,  96, 128,  9, 0, 128);  // 4 b2.fc1
    add(12, 0, 128,  96, 128,  96, 11, 0,  96);  // 5 b2.fc2
    add(26, 0,  96,  64,  96,  64, 25, 0,  64);  // 6 gate_l1
    add(28, 0,  64,  32,  64,  32, 27, 0,  32);  // 7 gate_l2
    add(30, 0,  32,   6,  32,  16, 29, 0,   6);  // 8 gate_l3 (N padded)
    for (int e = 0; e < 6; ++e) add(22, e * 9216, 96, 96, 96, 96, 17, e * 96, 96); //  9-14 w1
    for (int e = 0; e < 6; ++e) add(23, e * 9216, 96, 96, 96, 96, 18, e * 96, 96); // 15-20 w2
    for (int e = 0; e < 6; ++e) add(24, e * 9216, 96, 96, 96, 96, 19, e * 96, 96); // 21-26 w3
    add(50, 0, 96, 96, 96, 96, 49, 0, 96);       // 27 post_l
    add(54, 0, 96, 64, 96, 64, 53, 0, 64);       // 28 shared
    add(32, 0, 64, 64, 64, 64, 31, 0, 64);       // 29 strain l1
    add(34, 0, 64, 32, 64, 32, 33, 0, 32);       // 30 strain l2
    add(36, 0, 32,  1, 32, 16, 35, 0,  1);       // 31 strain l3
    add(38, 0, 64, 64, 64, 64, 37, 0, 64);       // 32 tensile l1
    add(40, 0, 64, 32, 64, 32, 39, 0, 32);       // 33 tensile l2
    add(42, 0, 32,  1, 32, 16, 41, 0,  1);       // 34 tensile l3
    add(64, 0, 64, 64, 64, 64, 63, 0, 64);       // 35 gap l1
    add(66, 0, 64, 32, 64, 32, 65, 0, 32);       // 36 gap l2
    add(68, 0, 32,  1, 32, 16, 67, 0,  1);       // 37 gap l3

    size_t wTot = 0, bTot = 0;
    size_t wOff[NLAYERS], bOff[NLAYERS];
    for (int i = 0; i < NLAYERS; ++i) { wOff[i] = wTot; wTot += (size_t)L[i].Kp * L[i].Np; }
    for (int i = 0; i < NLAYERS; ++i) { bOff[i] = bTot; bTot += (size_t)L[i].Np; }
    _Float16* wsW = (_Float16*)d_ws;
    float*    wsB = (float*)((char*)d_ws + wTot * sizeof(_Float16));

    KP p{};
    p.x   = (const float*)d_in[0];
    p.out = (float*)d_out;
    for (int i = 0; i < NLAYERS; ++i) {
        const float* srcW = (const float*)d_in[L[i].wi] + L[i].wo;
        const float* srcB = (const float*)d_in[L[i].bi] + L[i].bo;
        _Float16* dw = wsW + wOff[i];
        float*    db = wsB + bOff[i];
        int tot = L[i].Kp * L[i].Np;
        conv_w<<<(tot + 255) / 256, 256, 0, stream>>>(srcW, dw, L[i].K, L[i].N, L[i].Kp, L[i].Np);
        conv_b<<<1, 128, 0, stream>>>(srcB, db, L[i].bN, L[i].Np);
        p.w[i] = dw;
        p.b[i] = db;
    }
    p.g_stem1 = (const float*)d_in[60]; p.bt_stem1 = (const float*)d_in[59];
    p.g_stem2 = (const float*)d_in[62]; p.bt_stem2 = (const float*)d_in[61];
    p.g_b1l1  = (const float*)d_in[6];  p.bt_b1l1  = (const float*)d_in[5];
    p.g_b1l2  = (const float*)d_in[8];  p.bt_b1l2  = (const float*)d_in[7];
    p.g_b2l1  = (const float*)d_in[14]; p.bt_b2l1  = (const float*)d_in[13];
    p.g_b2l2  = (const float*)d_in[16]; p.bt_b2l2  = (const float*)d_in[15];
    p.e_lng   = (const float*)d_in[21]; p.e_lnb    = (const float*)d_in[20];
    p.g_post  = (const float*)d_in[52]; p.bt_post  = (const float*)d_in[51];

    const int B      = in_sizes[0] / 64;        // 262144 rows
    const int tiles  = B / 16;
    const int blocks = tiles / WAVES;
    fused_net<<<blocks, 32 * WAVES, 0, stream>>>(p);
}
